// CSINet_10642928959994
// MI455X (gfx1250) — matrix operand order repaired and checked by reading, not compiled, not run on hardware
//
#include <hip/hip_runtime.h>
#include <cstdint>

typedef _Float16 h16;
typedef __attribute__((ext_vector_type(16))) _Float16 v16h;
typedef __attribute__((ext_vector_type(8)))  float    v8f;

#define NOBJ   48
#define NPAIR  2256
#define HWSZ   196
#define KBIG   25088
#define TOT    2304   // NOBJ + NPAIR

// ---------------------------------------------------------------------------
// WMMA fragment loaders (gfx1250 16x16x32 f16 layouts, ISA 7.12.2)
// A (16x32): lane m = lane&15, khalf = lane>>4; elems 0..7 -> K=khalf*8+e,
//            elems 8..15 -> K=khalf*8+16+(e-8)  => two 16B loads.
// B (32x16): lane n = lane&15, khalf = lane>>4; elems 0..7 -> K=khalf*8+e,
//            elems 8..15 -> K=khalf*8+16+(e-8)  => two 16B loads.
// ---------------------------------------------------------------------------
union FragU { v16h v; uint4 q[2]; };

__device__ inline v16h load_fragA(const h16* rowptr, int kb, int khalf) {
  FragU f;
  const h16* p = rowptr + kb + khalf * 8;
  f.q[0] = *(const uint4*)(p);
  f.q[1] = *(const uint4*)(p + 16);
  return f.v;
}
__device__ inline v16h load_fragB(const h16* colptr, int kb, int khalf) {
  FragU f;
  const h16* p = colptr + kb + khalf * 8;
  f.q[0] = *(const uint4*)(p);
  f.q[1] = *(const uint4*)(p + 16);
  return f.v;
}

#define WMMA_F16(a, b, c) \
  __builtin_amdgcn_wmma_f32_16x16x32_f16(false, (a), false, (b), (short)0, (c), false, false)

// ---------------------------------------------------------------------------
// Generic WMMA GEMM:  C[M,N] = epi(A[M,K] @ Bt[N,K]^T)
//   A: f16 row-major (lda halves/row); Bt: f16 row n = K contiguous values.
//   MASK=true: A row scaled by rowmask[m][k>>7] (binary, hw-major K); mask
//              rows LDS-staged per block; all-zero 32-chunks skipped
//              wave-uniformly (saves loads and all 4 WMMAs).
//   MASK=false: register ping-pong pipeline (K % 64 == 0) overlapping
//               fragment loads with WMMA.
//   epilogue: +bias, relu, +residual(f32) -> Cf(f32) / Ch(f16) / ChT(f16^T)
// Block tile 64x128: 8 waves = 4 row-subtiles x 2 col-subtiles, 4 acc/wave.
// ---------------------------------------------------------------------------
template<bool MASK>
__global__ void __launch_bounds__(256) gemm_wmma(
    const h16* __restrict__ A, int lda,
    const h16* __restrict__ Bt,
    const float* __restrict__ bias,
    const float* __restrict__ residual, int ldres,
    float* __restrict__ Cf, int ldcf,
    h16* __restrict__ Ch, int ldch,
    h16* __restrict__ ChT, int ldct,
    int Mrows, int Ncols, int Npad, int K, int relu,
    const h16* __restrict__ rowmask)
{
  const int lane   = threadIdx.x & 31;
  const int wave   = threadIdx.x >> 5;
  const int rowsub = wave & 3;
  const int colsub = wave >> 2;
  const int l15    = lane & 15;
  const int khalf  = lane >> 4;

  const int mrow   = blockIdx.x * 64 + rowsub * 16 + l15;
  const int mclamp = (mrow < Mrows) ? mrow : (Mrows - 1);
  const h16* Arow  = A + (size_t)mclamp * lda;

  const h16* Bp[4];
  #pragma unroll
  for (int t = 0; t < 4; ++t) {
    int n = blockIdx.y * 128 + colsub * 64 + t * 16 + l15;
    Bp[t] = Bt + (size_t)((n < Npad) ? n : 0) * K;
  }

  v8f acc[4];
  #pragma unroll
  for (int t = 0; t < 4; ++t) acc[t] = (v8f){};

  if (MASK) {
    // Stage this block's 64 mask rows into LDS (25 KB) once; per-chunk mask
    // lookups then hit LDS instead of issuing 784 global_load_u16 per lane.
    __shared__ h16 lmask[64 * HWSZ];
    for (int idx = threadIdx.x; idx < 64 * HWSZ; idx += 256) {
      int rl = idx / HWSZ, cc = idx - rl * HWSZ;
      int gr = blockIdx.x * 64 + rl;
      int grc = (gr < Mrows) ? gr : (Mrows - 1);
      lmask[idx] = rowmask[(size_t)grc * HWSZ + cc];
    }
    __syncthreads();

    const h16* mrp = lmask + (size_t)(rowsub * 16 + l15) * HWSZ;
    for (int kb = 0; kb < K; kb += 32) {
      _Float16 mv = mrp[kb >> 7];
      if (__any((float)mv != 0.0f)) {          // wave-uniform chunk skip
        v16h a = load_fragA(Arow, kb, khalf) * mv;
        #pragma unroll
        for (int t = 0; t < 4; ++t) {
          v16h b = load_fragB(Bp[t], kb, khalf);
          acc[t] = WMMA_F16(a, b, acc[t]);
        }
      }
    }
  } else {
    // K is a multiple of 64 for all dense launches: ping-pong pipeline.
    v16h a0 = load_fragA(Arow, 0, khalf);
    v16h b0[4];
    #pragma unroll
    for (int t = 0; t < 4; ++t) b0[t] = load_fragB(Bp[t], 0, khalf);

    for (int kb = 0; kb < K; kb += 64) {
      v16h a1 = load_fragA(Arow, kb + 32, khalf);
      v16h b1[4];
      #pragma unroll
      for (int t = 0; t < 4; ++t) b1[t] = load_fragB(Bp[t], kb + 32, khalf);

      #pragma unroll
      for (int t = 0; t < 4; ++t) acc[t] = WMMA_F16(a0, b0[t], acc[t]);

      if (kb + 64 < K) {
        a0 = load_fragA(Arow, kb + 64, khalf);
        #pragma unroll
        for (int t = 0; t < 4; ++t) b0[t] = load_fragB(Bp[t], kb + 64, khalf);
        __builtin_prefetch(Arow + kb + 128, 0, 1);
      }
      #pragma unroll
      for (int t = 0; t < 4; ++t) acc[t] = WMMA_F16(a1, b1[t], acc[t]);
    }
  }

  // Epilogue: C element (row = r + 8*khalf in subtile, col = l15 in ntile)
  #pragma unroll
  for (int r = 0; r < 8; ++r) {
    int mr = blockIdx.x * 64 + rowsub * 16 + r + khalf * 8;
    if (mr >= Mrows) continue;
    #pragma unroll
    for (int t = 0; t < 4; ++t) {
      int nc = blockIdx.y * 128 + colsub * 64 + t * 16 + l15;
      if (nc >= Ncols) continue;
      float v = acc[t][r];
      if (bias)     v += bias[nc];
      if (relu)     v = (v > 0.0f) ? v : 0.0f;
      if (residual) v += residual[(size_t)mr * ldres + nc];
      if (Cf)  Cf[(size_t)mr * ldcf + nc] = v;
      if (Ch)  Ch[(size_t)mr * ldch + nc] = (h16)v;
      if (ChT) ChT[(size_t)nc * ldct + mr] = (h16)v;
    }
  }
}

// ---------------------------------------------------------------------------
// uf kernel: per pair m, uf2[m][hw][d] = sum_c U[m][c][hw]*Wc[d][c] + bc[d]
// WMMA: rows = hw (13 tiles), cols = d (8 waves, one d-tile each), K = c(256)
// Double-buffered LDS staging of U chunks + one-step register ping-pong on
// the A fragment so ds_load latency overlaps the WMMA pipe.
// ---------------------------------------------------------------------------
__global__ void __launch_bounds__(256) k_uf(
    const float* __restrict__ U, const h16* __restrict__ Wc16,
    const float* __restrict__ bc, h16* __restrict__ uf2)
{
  __shared__ __align__(16) h16 ldsU[2][208 * 32];
  const int m     = blockIdx.x;
  const int tid   = threadIdx.x;
  const int lane  = tid & 31;
  const int wave  = tid >> 5;
  const int l15   = lane & 15;
  const int khalf = lane >> 4;

  v8f acc[13];
  #pragma unroll
  for (int t = 0; t < 13; ++t) acc[t] = (v8f){};

  const h16* Bcol = Wc16 + (size_t)(wave * 16 + l15) * 256;

  auto stage = [&](int buf, int kc) {
    for (int idx = tid; idx < 32 * 208; idx += 256) {
      int c_local = idx / 208;
      int hwp = idx - c_local * 208;
      float v = 0.0f;
      if (hwp < HWSZ)
        v = U[((size_t)m * 256 + (kc + c_local)) * HWSZ + hwp];
      ldsU[buf][hwp * 32 + c_local] = (h16)v;
    }
  };

  stage(0, 0);
  __syncthreads();
  for (int i = 0; i < 8; ++i) {
    if (i + 1 < 8) stage((i + 1) & 1, (i + 1) * 32);
    const h16* cur = ldsU[i & 1];
    v16h b = load_fragB(Bcol, i * 32, khalf);

    // ping-pong A fragment: issue tile t+1's ds loads before tile t's WMMA
    v16h a_cur = load_fragA(cur + (size_t)l15 * 32, 0, khalf);
    #pragma unroll
    for (int t = 0; t < 13; ++t) {
      v16h a_nxt = a_cur;
      if (t < 12)
        a_nxt = load_fragA(cur + (size_t)((t + 1) * 16 + l15) * 32, 0, khalf);
      acc[t] = WMMA_F16(a_cur, b, acc[t]);
      a_cur = a_nxt;
    }
    __syncthreads();
  }

  const int d = wave * 16 + l15;
  const float biasd = bc[d];
  #pragma unroll
  for (int t = 0; t < 13; ++t) {
    #pragma unroll
    for (int r = 0; r < 8; ++r) {
      int hw = t * 16 + r + khalf * 8;
      if (hw < HWSZ)
        uf2[(size_t)m * KBIG + (size_t)hw * 128 + d] = (h16)(acc[t][r] + biasd);
    }
  }
}

// ---------------------------------------------------------------------------
// Rasterize sbj/obj/bg masks per pair (binary), store f16 planes [3][M][196]
// ---------------------------------------------------------------------------
__global__ void k_masks(const float* __restrict__ bboxes,
                        const int* __restrict__ pairs,
                        h16* __restrict__ maskbuf)
{
  const int m = blockIdx.x;
  const int s = pairs[2 * m], o = pairs[2 * m + 1];
  float sx1 = bboxes[s*4+0], sy1 = bboxes[s*4+1], sx2 = bboxes[s*4+2], sy2 = bboxes[s*4+3];
  float ox1 = bboxes[o*4+0], oy1 = bboxes[o*4+1], ox2 = bboxes[o*4+2], oy2 = bboxes[o*4+3];
  float ux = fminf(sx1, ox1), uy = fminf(sy1, oy1);
  float x0 = sx1-ux, x1 = sx2-ux, x2 = ox1-ux, x3 = ox2-ux;
  float y0 = sy1-uy, y1 = sy2-uy, y2 = oy1-uy, y3 = oy2-uy;
  float xf = 14.0f / fmaxf(x1, x3);
  float yf = 14.0f / fmaxf(y1, y3);
  int xp0 = (int)rintf(x0*xf), xp1 = (int)rintf(x1*xf);
  int xp2 = (int)rintf(x2*xf), xp3 = (int)rintf(x3*xf);
  int yp0 = (int)rintf(y0*yf), yp1 = (int)rintf(y1*yf);
  int yp2 = (int)rintf(y2*yf), yp3 = (int)rintf(y3*yf);
  for (int hw = threadIdx.x; hw < HWSZ; hw += blockDim.x) {
    int r = hw / 14, c = hw - 14 * (hw / 14);
    float sm = (r >= xp0 && r < xp1 && c >= yp0 && c < yp1) ? 1.0f : 0.0f;
    float om = (r >= xp2 && r < xp3 && c >= yp2 && c < yp3) ? 1.0f : 0.0f;
    float bm = fmaxf(1.0f - sm - om, 0.0f);
    maskbuf[((size_t)0 * NPAIR + m) * HWSZ + hw] = (h16)sm;
    maskbuf[((size_t)1 * NPAIR + m) * HWSZ + hw] = (h16)om;
    maskbuf[((size_t)2 * NPAIR + m) * HWSZ + hw] = (h16)bm;
  }
}

// ---------------------------------------------------------------------------
// Object-feature MLP (tiny, K=4251): hidden = relu(of@We1+be1); h0 = @We2+be2
// ---------------------------------------------------------------------------
__global__ void k_embed1(const float* __restrict__ roi, const float* __restrict__ logits,
                         const float* __restrict__ bb,  const float* __restrict__ We1,
                         const float* __restrict__ be1, float* __restrict__ hidden)
{
  const int n = blockIdx.x, j = threadIdx.x;
  float acc = be1[j];
  for (int k = 0; k < 4096; ++k) acc += roi[n*4096 + k]   * We1[(size_t)k * 128 + j];
  for (int k = 0; k < 151;  ++k) acc += logits[n*151 + k] * We1[(size_t)(4096 + k) * 128 + j];
  for (int k = 0; k < 4;    ++k) acc += bb[n*4 + k]       * We1[(size_t)(4247 + k) * 128 + j];
  hidden[n*128 + j] = fmaxf(acc, 0.0f);
}

__global__ void k_embed2(const float* __restrict__ hidden, const float* __restrict__ We2,
                         const float* __restrict__ be2, float* __restrict__ h,
                         h16* __restrict__ hh)
{
  const int n = blockIdx.x, j = threadIdx.x;
  float acc = be2[j];
  for (int k = 0; k < 128; ++k) acc += hidden[n*128 + k] * We2[k*128 + j];
  h[n*128 + j]  = acc;
  hh[n*128 + j] = (h16)acc;
}

// ---------------------------------------------------------------------------
// 3-token / 8-head attention per pair (tiny). One thread per (tq, d).
// ---------------------------------------------------------------------------
__global__ void k_attn(const float* __restrict__ qb, const float* __restrict__ kb,
                       const float* __restrict__ vb, h16* __restrict__ attn_out)
{
  const int m = blockIdx.x;
  const int t = threadIdx.x;        // 0..383
  const int tq = t >> 7;
  const int d = t & 127;
  const int hbase = d & ~15;
  const float* qp = qb + ((size_t)m * 3 + tq) * 128 + hbase;
  float s[3];
  #pragma unroll
  for (int tk = 0; tk < 3; ++tk) {
    const float* kp = kb + ((size_t)m * 3 + tk) * 128 + hbase;
    float dot = 0.0f;
    #pragma unroll
    for (int i = 0; i < 16; ++i) dot += qp[i] * kp[i];
    s[tk] = dot * 0.25f;            // 1/sqrt(16)
  }
  float mx = fmaxf(s[0], fmaxf(s[1], s[2]));
  float e0 = expf(s[0] - mx), e1 = expf(s[1] - mx), e2 = expf(s[2] - mx);
  float inv = 1.0f / (e0 + e1 + e2);
  float out = (e0 * vb[((size_t)m * 3 + 0) * 128 + d] +
               e1 * vb[((size_t)m * 3 + 1) * 128 + d] +
               e2 * vb[((size_t)m * 3 + 2) * 128 + d]) * inv;
  attn_out[((size_t)m * 3 + tq) * 128 + d] = (h16)out;
}

// ---------------------------------------------------------------------------
// Analytic adjacency (avoids materializing f32 adj): value of adj[i][j]
// ---------------------------------------------------------------------------
__device__ inline float adj_val(int i, int j, const int* __restrict__ pairs)
{
  if (i < NOBJ && j < NOBJ) return 1.0f;  // n2n covers all i!=j; +eye -> ones
  if (i < NOBJ) {
    int e = j - NOBJ;
    int s = pairs[2*e], o = pairs[2*e + 1];
    return (s == i || o == i) ? 1.0f : 0.0f;
  }
  if (j < NOBJ) {
    int e = i - NOBJ;
    int s = pairs[2*e], o = pairs[2*e + 1];
    return (s == j || o == j) ? 1.0f : 0.0f;
  }
  int e = i - NOBJ, f = j - NOBJ;
  if (e == f) return 1.0f;                // +eye
  int se = pairs[2*e], oe = pairs[2*e+1];
  int sf = pairs[2*f], of = pairs[2*f+1];
  int k1 = min(se, oe) * NOBJ + max(se, oe);
  int k2 = min(sf, of) * NOBJ + max(sf, of);
  return (k1 == k2) ? 1.0f : 0.0f;
}

__global__ void k_rowsum(const int* __restrict__ pairs, float* __restrict__ rowsum)
{
  __shared__ float red[256];
  const int i = blockIdx.x;
  float sum = 0.0f;
  for (int j = threadIdx.x; j < TOT; j += 256) sum += adj_val(i, j, pairs);
  red[threadIdx.x] = sum;
  __syncthreads();
  for (int s = 128; s > 0; s >>= 1) {
    if (threadIdx.x < s) red[threadIdx.x] += red[threadIdx.x + s];
    __syncthreads();
  }
  if (threadIdx.x == 0) rowsum[i] = red[0];
}

__global__ void k_adjfill(const int* __restrict__ pairs, const float* __restrict__ rowsum,
                          h16* __restrict__ adjn)
{
  size_t idx = (size_t)blockIdx.x * 256 + threadIdx.x;
  if (idx >= (size_t)TOT * TOT) return;
  int i = (int)(idx / TOT);
  int j = (int)(idx - (size_t)i * TOT);
  adjn[idx] = (h16)(adj_val(i, j, pairs) / rowsum[i]);
}

// ---------------------------------------------------------------------------
// Weight preparation
// ---------------------------------------------------------------------------
__global__ void k_f32toh16(const float* __restrict__ src, h16* __restrict__ dst, int n)
{
  int i = blockIdx.x * 256 + threadIdx.x;
  if (i < n) dst[i] = (h16)src[i];
}

// Wt[n][k] = W[k][n] (f32->f16), rows n >= N zero-padded (n < Npad)
__global__ void k_transpose(const float* __restrict__ W, h16* __restrict__ Wt,
                            int K, int N, int Npad)
{
  int idx = blockIdx.x * 256 + threadIdx.x;
  if (idx >= Npad * K) return;
  int n = idx / K, kk = idx - n * K;
  Wt[idx] = (n < N) ? (h16)W[(size_t)kk * N + n] : (h16)0.0f;
}

// Wt[n][hw*128 + d] = W[(d*196+hw)*128 + n]  (f32->f16, hw-major reorder)
__global__ void k_reorder25088(const float* __restrict__ W, h16* __restrict__ Wt)
{
  size_t idx = (size_t)blockIdx.x * 256 + threadIdx.x;
  if (idx >= (size_t)128 * KBIG) return;
  int n = (int)(idx / KBIG);
  int rem = (int)(idx - (size_t)n * KBIG);
  int hw = rem >> 7, d = rem & 127;
  Wt[idx] = (h16)W[((size_t)d * HWSZ + hw) * 128 + n];
}

// ---------------------------------------------------------------------------
extern "C" void kernel_launch(void* const* d_in, const int* in_sizes, int n_in,
                              void* d_out, int out_size, void* d_ws, size_t ws_size,
                              hipStream_t stream)
{
  (void)in_sizes; (void)n_in; (void)out_size; (void)ws_size;
  const float* roi    = (const float*)d_in[0];
  const float* logits = (const float*)d_in[1];
  const float* bboxes = (const float*)d_in[2];
  const float* U      = (const float*)d_in[3];
  const int*   pairs  = (const int*)  d_in[4];
  const float* We1 = (const float*)d_in[5];  const float* be1 = (const float*)d_in[6];
  const float* We2 = (const float*)d_in[7];  const float* be2 = (const float*)d_in[8];
  const float* Wc  = (const float*)d_in[9];  const float* bc  = (const float*)d_in[10];
  const float* Ws  = (const float*)d_in[11]; const float* bs  = (const float*)d_in[12];
  const float* Wob = (const float*)d_in[13]; const float* bob = (const float*)d_in[14];
  const float* Wbg = (const float*)d_in[15]; const float* bbg = (const float*)d_in[16];
  const float* Wq  = (const float*)d_in[17]; const float* bq  = (const float*)d_in[18];
  const float* Wk  = (const float*)d_in[19]; const float* bk  = (const float*)d_in[20];
  const float* Wv  = (const float*)d_in[21]; const float* bv  = (const float*)d_in[22];
  const float* Wo  = (const float*)d_in[23]; const float* bo  = (const float*)d_in[24];
  const float* Wr1 = (const float*)d_in[25]; const float* br1 = (const float*)d_in[26];
  const float* Wr2 = (const float*)d_in[27]; const float* br2 = (const float*)d_in[28];
  const float* gW  = (const float*)d_in[29]; const float* gb  = (const float*)d_in[30];
  const float* Wop = (const float*)d_in[31]; const float* bop = (const float*)d_in[32];
  const float* Wrp = (const float*)d_in[33]; const float* brp = (const float*)d_in[34];

  char* wsp = (char*)d_ws;
  size_t off = 0;
  auto wsalloc = [&](size_t bytes) -> void* {
    void* p = wsp + off;
    off += (bytes + 255) & ~(size_t)255;
    return p;
  };

  h16* Wc16  = (h16*)wsalloc((size_t)128 * 256 * 2);
  h16* WsT   = (h16*)wsalloc((size_t)128 * KBIG * 2);
  h16* WobT  = (h16*)wsalloc((size_t)128 * KBIG * 2);
  h16* WbgT  = (h16*)wsalloc((size_t)128 * KBIG * 2);
  h16* WqT   = (h16*)wsalloc((size_t)128 * 128 * 2);
  h16* WkT   = (h16*)wsalloc((size_t)128 * 128 * 2);
  h16* WvT   = (h16*)wsalloc((size_t)128 * 128 * 2);
  h16* WoT   = (h16*)wsalloc((size_t)128 * 128 * 2);
  h16* Wr1T  = (h16*)wsalloc((size_t)128 * 384 * 2);
  h16* Wr2T  = (h16*)wsalloc((size_t)128 * 128 * 2);
  h16* gWT   = (h16*)wsalloc((size_t)4 * 128 * 128 * 2);
  h16* WopT  = (h16*)wsalloc((size_t)160 * 128 * 2);
  h16* WrpT  = (h16*)wsalloc((size_t)64  * 128 * 2);
  h16* maskb = (h16*)wsalloc((size_t)3 * NPAIR * HWSZ * 2);
  h16* uf2   = (h16*)wsalloc((size_t)NPAIR * KBIG * 2);
  float* hidden = (float*)wsalloc((size_t)NOBJ * 128 * 4);
  h16* xseq  = (h16*)wsalloc((size_t)NPAIR * 3 * 128 * 2);
  float* qb  = (float*)wsalloc((size_t)NPAIR * 3 * 128 * 4);
  float* kbf = (float*)wsalloc((size_t)NPAIR * 3 * 128 * 4);
  float* vbf = (float*)wsalloc((size_t)NPAIR * 3 * 128 * 4);
  h16* att_o = (h16*)wsalloc((size_t)NPAIR * 3 * 128 * 2);
  h16* ocat  = (h16*)wsalloc((size_t)NPAIR * 3 * 128 * 2);   // viewed as [M][384]
  h16* r1buf = (h16*)wsalloc((size_t)NPAIR * 128 * 2);
  float* h   = (float*)wsalloc((size_t)TOT * 128 * 4);
  h16* hh    = (h16*)wsalloc((size_t)TOT * 128 * 2);
  h16* t1T   = (h16*)wsalloc((size_t)128 * TOT * 2);         // transposed h@W
  h16* adjn  = (h16*)wsalloc((size_t)TOT * TOT * 2);
  float* rsum = (float*)wsalloc((size_t)TOT * 4);

  float* obj_dists = (float*)d_out;            // [48,151]
  float* rel_dists = (float*)d_out + 48 * 151; // [2256,51]

  // ---- weight prep ----
  k_f32toh16<<<(128 * 256 + 255) / 256, 256, 0, stream>>>(Wc, Wc16, 128 * 256);
  {
    int g = (int)(((size_t)128 * KBIG + 255) / 256);
    k_reorder25088<<<g, 256, 0, stream>>>(Ws,  WsT);
    k_reorder25088<<<g, 256, 0, stream>>>(Wob, WobT);
    k_reorder25088<<<g, 256, 0, stream>>>(Wbg, WbgT);
  }
  k_transpose<<<(128 * 128 + 255) / 256, 256, 0, stream>>>(Wq, WqT, 128, 128, 128);
  k_transpose<<<(128 * 128 + 255) / 256, 256, 0, stream>>>(Wk, WkT, 128, 128, 128);
  k_transpose<<<(128 * 128 + 255) / 256, 256, 0, stream>>>(Wv, WvT, 128, 128, 128);
  k_transpose<<<(128 * 128 + 255) / 256, 256, 0, stream>>>(Wo, WoT, 128, 128, 128);
  k_transpose<<<(128 * 384 + 255) / 256, 256, 0, stream>>>(Wr1, Wr1T, 384, 128, 128);
  k_transpose<<<(128 * 128 + 255) / 256, 256, 0, stream>>>(Wr2, Wr2T, 128, 128, 128);
  for (int l = 0; l < 4; ++l)
    k_transpose<<<(128 * 128 + 255) / 256, 256, 0, stream>>>(
        gW + (size_t)l * 128 * 128, gWT + (size_t)l * 128 * 128, 128, 128, 128);
  k_transpose<<<(160 * 128 + 255) / 256, 256, 0, stream>>>(Wop, WopT, 128, 151, 160);
  k_transpose<<<(64  * 128 + 255) / 256, 256, 0, stream>>>(Wrp, WrpT, 128, 51, 64);

  // ---- object features ----
  k_embed1<<<NOBJ, 128, 0, stream>>>(roi, logits, bboxes, We1, be1, hidden);
  k_embed2<<<NOBJ, 128, 0, stream>>>(hidden, We2, be2, h, hh);

  // ---- masks + uf (f16, hw-major) ----
  k_masks<<<NPAIR, 224, 0, stream>>>(bboxes, pairs, maskb);
  k_uf<<<NPAIR, 256, 0, stream>>>(U, Wc16, bc, uf2);

  // ---- masked projections -> xseq [M][3][128] f16 ----
  dim3 gP((NPAIR + 63) / 64, 1);
  gemm_wmma<true><<<gP, 256, 0, stream>>>(uf2, KBIG, WsT,  bs,  nullptr, 0,
      nullptr, 0, xseq + 0,   384, nullptr, 0, NPAIR, 128, 128, KBIG, 0,
      maskb + (size_t)0 * NPAIR * HWSZ);
  gemm_wmma<true><<<gP, 256, 0, stream>>>(uf2, KBIG, WobT, bob, nullptr, 0,
      nullptr, 0, xseq + 128, 384, nullptr, 0, NPAIR, 128, 128, KBIG, 0,
      maskb + (size_t)1 * NPAIR * HWSZ);
  gemm_wmma<true><<<gP, 256, 0, stream>>>(uf2, KBIG, WbgT, bbg, nullptr, 0,
      nullptr, 0, xseq + 256, 384, nullptr, 0, NPAIR, 128, 128, KBIG, 0,
      maskb + (size_t)2 * NPAIR * HWSZ);

  // ---- QKV projections on [M*3][128] ----
  dim3 gQ((NPAIR * 3 + 63) / 64, 1);
  gemm_wmma<false><<<gQ, 256, 0, stream>>>(xseq, 128, WqT, bq, nullptr, 0,
      qb, 128, nullptr, 0, nullptr, 0, NPAIR * 3, 128, 128, 128, 0, nullptr);
  gemm_wmma<false><<<gQ, 256, 0, stream>>>(xseq, 128, WkT, bk, nullptr, 0,
      kbf, 128, nullptr, 0, nullptr, 0, NPAIR * 3, 128, 128, 128, 0, nullptr);
  gemm_wmma<false><<<gQ, 256, 0, stream>>>(xseq, 128, WvT, bv, nullptr, 0,
      vbf, 128, nullptr, 0, nullptr, 0, NPAIR * 3, 128, 128, 128, 0, nullptr);

  // ---- attention + output projection + rel MLP ----
  k_attn<<<NPAIR, 384, 0, stream>>>(qb, kbf, vbf, att_o);
  gemm_wmma<false><<<gQ, 256, 0, stream>>>(att_o, 128, WoT, bo, nullptr, 0,
      nullptr, 0, ocat, 128, nullptr, 0, NPAIR * 3, 128, 128, 128, 0, nullptr);
  gemm_wmma<false><<<gP, 256, 0, stream>>>(ocat, 384, Wr1T, br1, nullptr, 0,
      nullptr, 0, r1buf, 128, nullptr, 0, NPAIR, 128, 128, 384, 1, nullptr);
  gemm_wmma<false><<<gP, 256, 0, stream>>>(r1buf, 128, Wr2T, br2, nullptr, 0,
      h + 48 * 128, 128, hh + 48 * 128, 128, nullptr, 0, NPAIR, 128, 128, 128, 0, nullptr);

  // ---- normalized adjacency (f16) ----
  k_rowsum<<<TOT, 256, 0, stream>>>(pairs, rsum);
  k_adjfill<<<(int)(((size_t)TOT * TOT + 255) / 256), 256, 0, stream>>>(pairs, rsum, adjn);

  // ---- 4 GCN layers: h = relu(adjn @ (h @ W_l) + b_l) + h ----
  dim3 gG((TOT + 63) / 64, 1);
  for (int l = 0; l < 4; ++l) {
    gemm_wmma<false><<<gG, 256, 0, stream>>>(hh, 128, gWT + (size_t)l * 128 * 128,
        nullptr, nullptr, 0, nullptr, 0, nullptr, 0, t1T, TOT,
        TOT, 128, 128, 128, 0, nullptr);
    gemm_wmma<false><<<gG, 256, 0, stream>>>(adjn, TOT, t1T, gb + (size_t)l * 128,
        h, 128, h, 128, hh, 128, nullptr, 0, TOT, 128, 128, TOT, 1, nullptr);
  }

  // ---- classifier heads ----
  gemm_wmma<false><<<dim3(1, 2), 256, 0, stream>>>(hh, 128, WopT, bop, nullptr, 0,
      obj_dists, 151, nullptr, 0, nullptr, 0, 48, 151, 160, 128, 0, nullptr);
  gemm_wmma<false><<<gP, 256, 0, stream>>>(hh + 48 * 128, 128, WrpT, brp, nullptr, 0,
      rel_dists, 51, nullptr, 0, nullptr, 0, NPAIR, 51, 64, 128, 0, nullptr);
}
